// SpectralCOREL_66219805770147
// MI455X (gfx1250) — matrix operand order repaired and checked by reading, not compiled, not run on hardware
//
#include <hip/hip_runtime.h>
#include <hip/hip_bf16.h>
#include <cstdint>

// ---------------- problem constants (match reference) ----------------
#define Bn   128
#define Nn   283
#define En   4528
#define NE   (En + Nn)        // 4811 edges incl. self loops
#define Hh   4
#define Dd   64
#define HD   (Hh * Dd)        // 256
#define Pp   16
#define K1P  32               // IN1=17 padded to 32 for WMMA
#define BN   (Bn * Nn)        // 36224 rows (divisible by 16)

typedef _Float16 v8h  __attribute__((ext_vector_type(8)));
typedef _Float16 v16h __attribute__((ext_vector_type(16)));
typedef float    v8f  __attribute__((ext_vector_type(8)));

__device__ inline void edge_sd(const int* __restrict__ ei, int e, int& s, int& d) {
    if (e < En) { s = ei[e]; d = ei[En + e]; }
    else        { s = d = e - En; }
}
__device__ inline float lrelu02(float x) { return x >= 0.f ? x : 0.2f * x; }
__device__ inline float elu1(float x)    { return x > 0.f ? x : expm1f(x); }

// ---------------- pack weights to f16 (B-operand stored K-major) ----------------
__global__ void prep_weights(const float* __restrict__ W1, const float* __restrict__ W2,
                             _Float16* __restrict__ W1t, _Float16* __restrict__ W2t) {
    int tid = blockIdx.x * blockDim.x + threadIdx.x;
    if (tid < HD * K1P) {
        int col = tid >> 5, kk = tid & 31;
        float v = (kk < 17) ? W1[kk * HD + col] : 0.f;
        W1t[col * K1P + kk] = (_Float16)v;
    } else if (tid < HD * K1P + Dd * HD) {
        int t2 = tid - HD * K1P;
        int n = t2 >> 8, k = t2 & 255;
        W2t[n * HD + k] = (_Float16)W2[k * Dd + n];
    }
}

// x1h[r][c] f16: c==0 -> sigma, 1..16 -> pos_emb[n], 17..31 -> 0
__global__ void prep_x1(const float* __restrict__ sigma, const float* __restrict__ pos,
                        _Float16* __restrict__ x1h) {
    int tid = blockIdx.x * blockDim.x + threadIdx.x;
    if (tid >= BN * K1P) return;
    int r = tid >> 5, c = tid & 31;
    int n = r % Nn;
    float v = (c == 0) ? sigma[r] : ((c <= Pp) ? pos[n * Pp + (c - 1)] : 0.f);
    x1h[tid] = (_Float16)v;
}

// ---------------- CSR of incoming edges (deterministic, stable order) ----------
// Count in-degrees (dst list staged in LDS), serial prefix -> off[0..Nn]
__global__ void csr_count(const int* __restrict__ ei, int* __restrict__ off) {
    __shared__ int dstL[NE];
    __shared__ int degL[Nn];
    for (int i = threadIdx.x; i < NE; i += blockDim.x) {
        int s, d; edge_sd(ei, i, s, d); (void)s;
        dstL[i] = d;
    }
    __syncthreads();
    for (int n = threadIdx.x; n < Nn; n += blockDim.x) {
        int c = 0;
        for (int j = 0; j < NE; ++j) c += (dstL[j] == n);
        degL[n] = c;
    }
    __syncthreads();
    if (threadIdx.x == 0) {
        int acc = 0;
        for (int n = 0; n < Nn; ++n) { off[n] = acc; acc += degL[n]; }
        off[Nn] = acc;
    }
}

// One block per dst node; wave32 ballot-compaction preserves edge order.
__global__ void csr_fill(const int* __restrict__ ei, const int* __restrict__ off,
                         int* __restrict__ srcs) {
    __shared__ int dstL[NE];
    __shared__ int srcL[NE];
    const int n = blockIdx.x;
    for (int i = threadIdx.x; i < NE; i += blockDim.x) {
        int s, d; edge_sd(ei, i, s, d);
        srcL[i] = s; dstL[i] = d;
    }
    __syncthreads();
    if (threadIdx.x < 32) {
        const int lane = threadIdx.x;
        const int base0 = off[n];
        int cnt = 0;
        for (int base = 0; base < NE; base += 32) {
            int j = base + lane;
            bool pred = (j < NE) && (dstL[j] == n);
            unsigned long long mask = __ballot(pred);
            if (pred) {
                int pos = cnt + (int)__popcll(mask & ((1ull << lane) - 1ull));
                srcs[base0 + pos] = srcL[j];
            }
            cnt += (int)__popcll(mask);
        }
    }
}

// ---------------- WMMA GEMM: C[M,Nout] = A[M,K] * Bt[Nout,K]^T ----------------
// One wave computes a 16x64 strip: A fragment loaded once per k-step, 4 WMMAs.
// Per-lane layouts per CDNA5 ISA 7.12.2:
//   A: lane l -> row (l&15); K groups of 8 at {kb, 16+kb}, kb = 8*(l>>4)
//   B: lane l -> col (l&15); K groups of 8 at {kb, 16+kb}, kb = 8*(l>>4)
//   C: lane l, vgpr i -> row i + 8*(l>>4), col (l&15)
template <int K>
__global__ __launch_bounds__(128)
void wmma_gemm4(const _Float16* __restrict__ A, const _Float16* __restrict__ Bt,
                float* __restrict__ C, int M, int Nout) {
    const int lane = threadIdx.x & 31;
    const int wave = threadIdx.x >> 5;
    const int mt   = blockIdx.x * 4 + wave;
    const int nt0  = blockIdx.y * 4;
    if (mt >= (M >> 4)) return;                 // uniform per wave (grid divides exactly)
    const int hi  = lane >> 4;
    const int l15 = lane & 15;
    const int kbA = hi << 3;                    // 0 | 8
    const int kbB = hi << 3;                    // 0 | 8
    const _Float16* Arow = A  + (size_t)((mt << 4) + l15) * K;
    const _Float16* Brow = Bt + (size_t)((nt0 << 4) + l15) * K;

    v8f acc[4] = {};
#pragma unroll
    for (int k0 = 0; k0 < K; k0 += 32) {
        v8h alo = *(const v8h*)(Arow + k0 + kbA);
        v8h ahi = *(const v8h*)(Arow + k0 + kbA + 16);
        v16h a = __builtin_shufflevector(alo, ahi, 0,1,2,3,4,5,6,7,8,9,10,11,12,13,14,15);
#pragma unroll
        for (int t = 0; t < 4; ++t) {
            const _Float16* bp = Brow + (size_t)t * 16 * K + k0 + kbB;
            v8h blo = *(const v8h*)(bp);
            v8h bhi = *(const v8h*)(bp + 16);
            v16h b = __builtin_shufflevector(blo, bhi, 0,1,2,3,4,5,6,7,8,9,10,11,12,13,14,15);
            acc[t] = __builtin_amdgcn_wmma_f32_16x16x32_f16(
                         false, a, false, b, (short)0, acc[t], false, false);
        }
    }
#pragma unroll
    for (int t = 0; t < 4; ++t) {
        float* Cc = C + (size_t)(mt << 4) * Nout + ((nt0 + t) << 4) + l15;
#pragma unroll
        for (int i = 0; i < 8; ++i)
            Cc[(size_t)(i + (hi << 3)) * Nout] = acc[t][i];
    }
}

// ---------------- attention logits: al[r,h] = sum_d h[r,h,d]*a[h,d] ----------------
__global__ void logits_kernel(const float* __restrict__ h, const float* __restrict__ as,
                              const float* __restrict__ ad, float* __restrict__ als,
                              float* __restrict__ ald, int H) {
    int idx = blockIdx.x * blockDim.x + threadIdx.x;
    if (idx >= BN * H) return;
    int r = idx / H, hh = idx % H;
    const float* hp = h + (size_t)r * H * Dd + hh * Dd;
    float ss = 0.f, sd = 0.f;
#pragma unroll 8
    for (int d = 0; d < Dd; ++d) {
        float v = hp[d];
        ss = fmaf(v, as[hh * Dd + d], ss);
        sd = fmaf(v, ad[hh * Dd + d], sd);
    }
    als[idx] = ss;
    ald[idx] = sd;
}

// ---------------- per-dst softmax (gather, no atomics) ----------------
// thread per (n,b,h); writes NORMALIZED alpha at CSR position p: alpha[p][b][h]
__global__ void softmax_gather(const int* __restrict__ off, const int* __restrict__ srcs,
                               const float* __restrict__ als, const float* __restrict__ ald,
                               float* __restrict__ alpha, int H) {
    int idx = blockIdx.x * blockDim.x + threadIdx.x;
    if (idx >= Nn * Bn * H) return;
    int hh = idx % H;
    int t  = idx / H;
    int b  = t % Bn;
    int n  = t / Bn;
    int o0 = off[n], o1 = off[n + 1];
    float aldv = ald[(b * Nn + n) * H + hh];
    float mx = -__builtin_huge_valf();
    for (int p = o0; p < o1; ++p) {
        float lg = lrelu02(als[(b * Nn + srcs[p]) * H + hh] + aldv);
        mx = fmaxf(mx, lg);
    }
    float den = 0.f;
    for (int p = o0; p < o1; ++p) {
        float lg = lrelu02(als[(b * Nn + srcs[p]) * H + hh] + aldv);
        den += expf(lg - mx);
    }
    float inv = 1.f / den;
    for (int p = o0; p < o1; ++p) {
        float lg = lrelu02(als[(b * Nn + srcs[p]) * H + hh] + aldv);
        alpha[(size_t)p * Bn * H + b * H + hh] = expf(lg - mx) * inv;
    }
}

// ---------------- fused message gather + bias + ELU ----------------
// block (n, b), C threads (channels); edge ids + alphas staged in LDS tiles.
template <int C, int H>
__global__ __launch_bounds__(C)
void msg_gather(const int* __restrict__ off, const int* __restrict__ srcs,
                const float* __restrict__ alpha, const float* __restrict__ hlin,
                const float* __restrict__ bias, _Float16* __restrict__ o16,
                float* __restrict__ o32) {
    __shared__ int   sSrc[64];
    __shared__ float sAl[64 * H];
    const int n = blockIdx.x, b = blockIdx.y;
    const int c = threadIdx.x;
    const int hh = (H == 1) ? 0 : (c >> 6);
    const int o0 = off[n];
    const int dn = off[n + 1] - o0;

    float acc = 0.f;
    for (int t0 = 0; t0 < dn; t0 += 64) {
        int nt = dn - t0; if (nt > 64) nt = 64;
        for (int i = c; i < nt; i += C) sSrc[i] = srcs[o0 + t0 + i];
        for (int i = c; i < nt * H; i += C)
            sAl[i] = alpha[(size_t)(o0 + t0 + i / H) * Bn * H + b * H + (i % H)];
        __syncthreads();
        for (int j = 0; j < nt; ++j)
            acc = fmaf(sAl[j * H + hh], hlin[((size_t)b * Nn + sSrc[j]) * C + c], acc);
        __syncthreads();
    }
    float v = elu1(acc + bias[c]);
    size_t oidx = ((size_t)b * Nn + n) * C + c;
    if (o16) o16[oidx] = (_Float16)v;
    if (o32) o32[oidx] = v;
}

// ---------------- heads ----------------
__global__ void heads_kernel(const float* __restrict__ hfin, const float* __restrict__ mu_pred,
                             const float* __restrict__ Wm, const float* __restrict__ bm,
                             const float* __restrict__ Wr, const float* __restrict__ br,
                             float* __restrict__ out) {
    int r = blockIdx.x * blockDim.x + threadIdx.x;
    if (r >= BN) return;
    const float* hp = hfin + (size_t)r * Dd;
    float sm = 0.f, sr = 0.f;
#pragma unroll 8
    for (int d = 0; d < Dd; ++d) {
        float v = hp[d];
        sm = fmaf(v, Wm[d], sm);
        sr = fmaf(v, Wr[d], sr);
    }
    out[r] = mu_pred[r] + sm + bm[0];
    float x = sr + br[0];
    out[BN + r] = (x > 20.f) ? x : log1pf(expf(x));
}

// ---------------- host side ----------------
static inline size_t bump(size_t& off, size_t bytes) {
    size_t o = off;
    off = (off + bytes + 255) & ~(size_t)255;
    return o;
}

extern "C" void kernel_launch(void* const* d_in, const int* in_sizes, int n_in,
                              void* d_out, int out_size, void* d_ws, size_t ws_size,
                              hipStream_t stream) {
    const float* mu_pred = (const float*)d_in[0];
    const float* sigma   = (const float*)d_in[1];
    const int*   ei      = (const int*)  d_in[2];
    const float* pos     = (const float*)d_in[3];
    const float* W1      = (const float*)d_in[4];
    const float* a1s     = (const float*)d_in[5];
    const float* a1d     = (const float*)d_in[6];
    const float* b1      = (const float*)d_in[7];
    const float* W2      = (const float*)d_in[8];
    const float* a2s     = (const float*)d_in[9];
    const float* a2d     = (const float*)d_in[10];
    const float* b2      = (const float*)d_in[11];
    const float* Wm      = (const float*)d_in[12];
    const float* bm      = (const float*)d_in[13];
    const float* Wr      = (const float*)d_in[14];
    const float* br      = (const float*)d_in[15];
    float* out = (float*)d_out;
    char* ws = (char*)d_ws;

    size_t off = 0;
    _Float16* x1h    = (_Float16*)(ws + bump(off, (size_t)BN * K1P * 2));
    _Float16* W1t    = (_Float16*)(ws + bump(off, (size_t)HD * K1P * 2));
    _Float16* W2t    = (_Float16*)(ws + bump(off, (size_t)Dd * HD * 2));
    float*    hlin1  = (float*)   (ws + bump(off, (size_t)BN * HD * 4));
    float*    al1s   = (float*)   (ws + bump(off, (size_t)BN * Hh * 4));
    float*    al1d   = (float*)   (ws + bump(off, (size_t)BN * Hh * 4));
    float*    alpha1 = (float*)   (ws + bump(off, (size_t)NE * Bn * Hh * 4));
    _Float16* x2h    = (_Float16*)(ws + bump(off, (size_t)BN * HD * 2));
    float*    hlin2  = (float*)   (ws + bump(off, (size_t)BN * Dd * 4));
    float*    al2s   = (float*)   (ws + bump(off, (size_t)BN * 4));
    float*    al2d   = (float*)   (ws + bump(off, (size_t)BN * 4));
    float*    alpha2 = (float*)   (ws + bump(off, (size_t)NE * Bn * 4));
    float*    hfin   = (float*)   (ws + bump(off, (size_t)BN * Dd * 4));
    int*      offv   = (int*)     (ws + bump(off, (size_t)(Nn + 1) * 4));
    int*      srcs   = (int*)     (ws + bump(off, (size_t)NE * 4));

    auto blocks = [](long long n, int bs) { return (unsigned)((n + bs - 1) / bs); };

    // ---- prep: f16 operands + CSR of incoming edges ----
    prep_weights<<<blocks(HD * K1P + Dd * HD, 256), 256, 0, stream>>>(W1, W2, W1t, W2t);
    prep_x1<<<blocks((long long)BN * K1P, 256), 256, 0, stream>>>(sigma, pos, x1h);
    csr_count<<<1, 512, 0, stream>>>(ei, offv);
    csr_fill<<<Nn, 256, 0, stream>>>(ei, offv, srcs);

    // ---- layer 1 ----
    {
        dim3 g((BN / 16) / 4, HD / 64);
        wmma_gemm4<K1P><<<g, 128, 0, stream>>>(x1h, W1t, hlin1, BN, HD);
    }
    logits_kernel<<<blocks((long long)BN * Hh, 256), 256, 0, stream>>>(hlin1, a1s, a1d, al1s, al1d, Hh);
    softmax_gather<<<blocks((long long)Nn * Bn * Hh, 256), 256, 0, stream>>>(offv, srcs, al1s, al1d, alpha1, Hh);
    {
        dim3 g(Nn, Bn);
        msg_gather<HD, Hh><<<g, HD, 0, stream>>>(offv, srcs, alpha1, hlin1, b1, x2h, nullptr);
    }

    // ---- layer 2 ----
    {
        dim3 g((BN / 16) / 4, Dd / 64);
        wmma_gemm4<HD><<<g, 128, 0, stream>>>(x2h, W2t, hlin2, BN, Dd);
    }
    logits_kernel<<<blocks(BN, 256), 256, 0, stream>>>(hlin2, a2s, a2d, al2s, al2d, 1);
    softmax_gather<<<blocks((long long)Nn * Bn, 256), 256, 0, stream>>>(offv, srcs, al2s, al2d, alpha2, 1);
    {
        dim3 g(Nn, Bn);
        msg_gather<Dd, 1><<<g, Dd, 0, stream>>>(offv, srcs, alpha2, hlin2, b2, nullptr, hfin);
    }

    // ---- heads ----
    heads_kernel<<<blocks(BN, 256), 256, 0, stream>>>(hfin, mu_pred, Wm, bm, Wr, br, out);
}
